// VariationalGNN_38354057953764
// MI455X (gfx1250) — matrix operand order repaired and checked by reading, not compiled, not run on hardware
//
#include <hip/hip_runtime.h>

// ---------------------------------------------------------------------------
// Problem constants (match reference)
// ---------------------------------------------------------------------------
constexpr int Dd = 256;          // feature dim
constexpr int Nn = 25600;        // total nodes (B*P)
constexpr int Bb = 128;          // graphs
constexpr int Pp = 200;          // nodes per graph
constexpr int Ee = 409600;       // encode edges (before self loops)
constexpr float LN_EPS = 1e-5f;
constexpr float MAX_LOGSTD = 10.0f;

// ---------------------------------------------------------------------------
// CDNA5 WMMA types (wave32)
// ---------------------------------------------------------------------------
typedef __attribute__((ext_vector_type(16))) __bf16 v16bf;
typedef __attribute__((ext_vector_type(8)))  __bf16 bf16x8;
typedef __attribute__((ext_vector_type(8)))  float  v8f;
typedef int v4i __attribute__((vector_size(16)));

// Async global->LDS copy (gfx1250): use the clang builtins when available,
// otherwise fall back to plain load/store staging (still compiles everywhere).
#if defined(__gfx1250__) && __has_builtin(__builtin_amdgcn_global_load_async_to_lds_b128) && __has_builtin(__builtin_amdgcn_s_wait_asynccnt)
#define USE_ASYNC_LDS 1
typedef __attribute__((address_space(1))) v4i* gv4i_p;  // global int4*
typedef __attribute__((address_space(3))) v4i* lv4i_p;  // LDS int4*
#else
#define USE_ASYNC_LDS 0
#endif

// ---------------------------------------------------------------------------
// Helpers
// ---------------------------------------------------------------------------
__device__ __forceinline__ float wave_sum32(float v) {
#pragma unroll
  for (int off = 16; off > 0; off >>= 1) v += __shfl_down(v, off, 32);
  return __shfl(v, 0, 32);
}

__device__ __forceinline__ void atomicMaxFloat(float* addr, float val) {
  // classic bit-trick: works with init pattern 0xFF800000 (-inf)
  if (val >= 0.0f)
    atomicMax(reinterpret_cast<int*>(addr), __float_as_int(val));
  else
    atomicMin(reinterpret_cast<unsigned int*>(addr),
              static_cast<unsigned int>(__float_as_int(val)));
}

// ---------------------------------------------------------------------------
// Setup kernels
// ---------------------------------------------------------------------------
// fp32 W[K x Ncols] (row-major) -> bf16 Wt[Ncols x K] (transposed, row-major)
__global__ void k_w_to_bf16_t(const float* __restrict__ W, __bf16* __restrict__ Wt,
                              int K, int Ncols) {
  int idx = blockIdx.x * blockDim.x + threadIdx.x;
  if (idx >= K * Ncols) return;
  int k = idx / Ncols, n = idx % Ncols;
  Wt[(size_t)n * K + k] = (__bf16)W[idx];
}

// embedding gather -> bf16 activations
__global__ void k_gather(const int* __restrict__ idx, const float* __restrict__ emb,
                         __bf16* __restrict__ xb) {
  int i = blockIdx.x * blockDim.x + threadIdx.x;
  if (i >= Nn * Dd) return;
  int n = i / Dd, d = i % Dd;
  xb[i] = (__bf16)emb[(size_t)idx[n] * Dd + d];
}

__global__ void k_init_softmax(float* __restrict__ emax, float* __restrict__ den) {
  int i = blockIdx.x * blockDim.x + threadIdx.x;
  if (i >= Nn) return;
  emax[i] = __int_as_float(0xFF800000);  // -inf
  den[i] = 0.0f;
}

__global__ void k_init_out_bias(float* __restrict__ out, const float* __restrict__ bias) {
  int i = blockIdx.x * blockDim.x + threadIdx.x;
  if (i >= Nn * Dd) return;
  out[i] = bias[i % Dd];
}

__global__ void k_zero1(float* p) { *p = 0.0f; }

// combined edge list: nE_base edges from `pair` (+ Nn self loops)
__global__ void k_build_edges(const int* __restrict__ pair, int nE_base,
                              int* __restrict__ src, int* __restrict__ dst) {
  int i = blockIdx.x * blockDim.x + threadIdx.x;
  int tot = nE_base + Nn;
  if (i >= tot) return;
  if (i < nE_base) { src[i] = pair[i]; dst[i] = pair[nE_base + i]; }
  else             { int v = i - nE_base; src[i] = v; dst[i] = v; }
}

// ---------------------------------------------------------------------------
// bf16 WMMA GEMM:  C[M x Ncols] = A[M x K] * Wt^T + bias       (K == 256)
//
// Block = 128 threads = 4 waves, computes a 64x64 output tile.
//   * B tile (64 cols x 256 K of bf16 = 32 KB, contiguous in Wt layout) is
//     staged once into LDS via async global->LDS b128 copies (ASYNCcnt),
//     then shared by all 4 waves (ds_load_b128 fragment reads).
//   * Each wave register-blocks 16 rows x 64 cols: the A fragment is loaded
//     once per k-step and reused across 4 v_wmma_f32_16x16x32_bf16.
// Grid is exact (M%64==0, Ncols%64==0) so EXEC is all-ones throughout.
// ---------------------------------------------------------------------------
__global__ void __launch_bounds__(128)
k_wmma_gemm(const __bf16* __restrict__ A, const __bf16* __restrict__ Wt,
            const float* __restrict__ bias,
            float* __restrict__ Cf, __bf16* __restrict__ Cb,
            int M, int Ncols, int K) {
  __shared__ __bf16 Bsh[64 * 256];  // 32 KB

  int wid = threadIdx.x >> 5;
  int lane = threadIdx.x & 31;
  int lane16 = lane & 15, hi = lane >> 4;
  int tilesN = Ncols >> 6;
  int bm = blockIdx.x / tilesN, bn = blockIdx.x % tilesN;

  // ---- stage the 64x256 B tile into LDS (contiguous 32 KB in Wt layout) ----
  const __bf16* gB = Wt + (size_t)bn * 64 * 256;
#if USE_ASYNC_LDS
#pragma unroll
  for (int i = 0; i < 16; i++) {
    int c = threadIdx.x + i * 128;  // 2048 chunks of 16B
    __builtin_amdgcn_global_load_async_to_lds_b128(
        (gv4i_p)(void*)const_cast<__bf16*>(gB + c * 8),
        (lv4i_p)(void*)(&Bsh[c * 8]), 0, 0);
  }
  __builtin_amdgcn_s_wait_asynccnt(0);
#else
#pragma unroll
  for (int i = 0; i < 16; i++) {
    int c = threadIdx.x + i * 128;
    *(float4*)(&Bsh[c * 8]) = *(const float4*)(gB + c * 8);
  }
#endif
  __syncthreads();

  // ---- register-blocked 16x64 per wave ----
  const __bf16* arow = A + (size_t)(bm * 64 + wid * 16 + lane16) * K;
  v8f acc[4] = {v8f{}, v8f{}, v8f{}, v8f{}};

  for (int k0 = 0; k0 < 256; k0 += 32) {
    bf16x8 a0 = *(const bf16x8*)(arow + k0 + hi * 8);
    bf16x8 a1 = *(const bf16x8*)(arow + k0 + 16 + hi * 8);
    v16bf a;
#pragma unroll
    for (int i = 0; i < 8; i++) { a[i] = a0[i]; a[8 + i] = a1[i]; }
#pragma unroll
    for (int t = 0; t < 4; t++) {
      const __bf16* bp = &Bsh[(t * 16 + lane16) * 256 + k0 + hi * 8];
      bf16x8 b0 = *(const bf16x8*)bp;
      bf16x8 b1 = *(const bf16x8*)(bp + 16);
      v16bf b;
#pragma unroll
      for (int i = 0; i < 8; i++) { b[i] = b0[i]; b[8 + i] = b1[i]; }
      acc[t] = __builtin_amdgcn_wmma_f32_16x16x32_bf16(false, a, false, b,
                                                       (short)0, acc[t],
                                                       false, false);
    }
  }

  // ---- store (C layout: lane16 = N, vgpr r -> M = r + 8*hi) ----
#pragma unroll
  for (int t = 0; t < 4; t++) {
    int n = bn * 64 + t * 16 + lane16;
    float bv = bias ? bias[n] : 0.0f;
#pragma unroll
    for (int r = 0; r < 8; r++) {
      int m = bm * 64 + wid * 16 + r + hi * 8;
      float v = acc[t][r] + bv;
      if (Cf) Cf[(size_t)m * Ncols + n] = v;
      if (Cb) Cb[(size_t)m * Ncols + n] = (__bf16)v;
    }
  }
}

// ---------------------------------------------------------------------------
// GATv2 edge passes (wave per edge; lane covers 8 contiguous features)
// ---------------------------------------------------------------------------
__global__ void k_edge_score(const int* __restrict__ src, const int* __restrict__ dst,
                             const float* __restrict__ xl, const float* __restrict__ xr,
                             const float* __restrict__ att,
                             float* __restrict__ e, float* __restrict__ emax, int nE) {
  int wid = blockIdx.x * (blockDim.x >> 5) + (threadIdx.x >> 5);
  int lane = threadIdx.x & 31;
  if (wid >= nE) return;
  int s = src[wid], t = dst[wid];
  const float* pl = xl + (size_t)s * Dd + lane * 8;
  const float* pr = xr + (size_t)t * Dd + lane * 8;
  const float* pa = att + lane * 8;
  float acc = 0.0f;
#pragma unroll
  for (int j = 0; j < 8; j++) {
    float v = pl[j] + pr[j];
    v = v > 0.0f ? v : 0.2f * v;  // leaky_relu(., 0.2)
    acc += v * pa[j];
  }
  acc = wave_sum32(acc);
  if (lane == 0) { e[wid] = acc; atomicMaxFloat(emax + t, acc); }
}

__global__ void k_edge_exp(const int* __restrict__ dst, const float* __restrict__ e,
                           const float* __restrict__ emax, float* __restrict__ den,
                           float* __restrict__ ex, int nE) {
  int i = blockIdx.x * blockDim.x + threadIdx.x;
  if (i >= nE) return;
  int t = dst[i];
  float v = __expf(e[i] - emax[t]);
  ex[i] = v;
  atomicAdd(den + t, v);
}

__global__ void k_edge_aggr(const int* __restrict__ src, const int* __restrict__ dst,
                            const float* __restrict__ xl, const float* __restrict__ ex,
                            const float* __restrict__ den, float* __restrict__ out,
                            int nE) {
  int wid = blockIdx.x * (blockDim.x >> 5) + (threadIdx.x >> 5);
  int lane = threadIdx.x & 31;
  if (wid >= nE) return;
  int s = src[wid], t = dst[wid];
  float w = ex[wid] / den[t];
  const float* pl = xl + (size_t)s * Dd + lane * 8;
  float* po = out + (size_t)t * Dd + lane * 8;
#pragma unroll
  for (int j = 0; j < 8; j++) atomicAdd(po + j, w * pl[j]);
}

// ---------------------------------------------------------------------------
// LayerNorm (+ReLU) : one wave per row, writes bf16 for the next WMMA GEMM
// ---------------------------------------------------------------------------
__global__ void k_ln_relu(const float* __restrict__ X, const float* __restrict__ g,
                          const float* __restrict__ b, __bf16* __restrict__ Yb, int M) {
  int wid = blockIdx.x * (blockDim.x >> 5) + (threadIdx.x >> 5);
  int lane = threadIdx.x & 31;
  if (wid >= M) return;
  const float* px = X + (size_t)wid * Dd + lane * 8;
  float v[8], s = 0.0f;
#pragma unroll
  for (int j = 0; j < 8; j++) { v[j] = px[j]; s += v[j]; }
  float mean = wave_sum32(s) * (1.0f / Dd);
  float sq = 0.0f;
#pragma unroll
  for (int j = 0; j < 8; j++) { float d = v[j] - mean; sq += d * d; }
  float var = wave_sum32(sq) * (1.0f / Dd);
  float inv = rsqrtf(var + LN_EPS);
  __bf16* py = Yb + (size_t)wid * Dd + lane * 8;
  const float* pg = g + lane * 8;
  const float* pb = b + lane * 8;
#pragma unroll
  for (int j = 0; j < 8; j++) {
    float y = (v[j] - mean) * inv * pg[j] + pb[j];
    py[j] = (__bf16)(y > 0.0f ? y : 0.0f);
  }
}

// mu (bf16) + per-node KL, accumulated into the kld output scalar.
// counts are uniformly P, so kld = (0.5/P) * sum_n kl_node[n].
__global__ void k_mu_kl(const float* __restrict__ h, __bf16* __restrict__ mub,
                        float* __restrict__ kl_out) {
  int wid = blockIdx.x * (blockDim.x >> 5) + (threadIdx.x >> 5);
  int lane = threadIdx.x & 31;
  if (wid >= Nn) return;
  const float* ph = h + (size_t)wid * (2 * Dd);
  float kl = 0.0f;
#pragma unroll
  for (int j = 0; j < 8; j++) {
    int d = lane * 8 + j;
    float mu = ph[d];
    float lv = fminf(ph[Dd + d], MAX_LOGSTD);
    mub[(size_t)wid * Dd + d] = (__bf16)mu;
    kl += __expf(lv) - lv - 1.0f + mu * mu;
  }
  kl = wave_sum32(kl);
  if (lane == 0) atomicAdd(kl_out, kl * (0.5f / (float)Pp));
}

// LN+ReLU on the B selected readout rows (last node of each graph)
__global__ void k_ln_rows(const float* __restrict__ X, const float* __restrict__ g,
                          const float* __restrict__ b, float* __restrict__ xg) {
  int wid = blockIdx.x * (blockDim.x >> 5) + (threadIdx.x >> 5);
  int lane = threadIdx.x & 31;
  if (wid >= Bb) return;
  int row = (wid + 1) * Pp - 1;
  const float* px = X + (size_t)row * Dd + lane * 8;
  float v[8], s = 0.0f;
#pragma unroll
  for (int j = 0; j < 8; j++) { v[j] = px[j]; s += v[j]; }
  float mean = wave_sum32(s) * (1.0f / Dd);
  float sq = 0.0f;
#pragma unroll
  for (int j = 0; j < 8; j++) { float d = v[j] - mean; sq += d * d; }
  float var = wave_sum32(sq) * (1.0f / Dd);
  float inv = rsqrtf(var + LN_EPS);
#pragma unroll
  for (int j = 0; j < 8; j++) {
    int d = lane * 8 + j;
    float y = (v[j] - mean) * inv * g[d] + b[d];
    xg[(size_t)wid * Dd + d] = y > 0.0f ? y : 0.0f;
  }
}

// num = relu(ln(demo @ num_W)) : one block (128 thr) per graph
__global__ void k_num(const float* __restrict__ demo, const float* __restrict__ W,
                      const float* __restrict__ g, const float* __restrict__ b,
                      float* __restrict__ num) {
  __shared__ float red[128];
  int gb = blockIdx.x, j = threadIdx.x;
  float acc = 0.0f;
#pragma unroll
  for (int k = 0; k < 16; k++) acc += demo[gb * 16 + k] * W[k * 128 + j];
  red[j] = acc; __syncthreads();
  for (int s = 64; s > 0; s >>= 1) { if (j < s) red[j] += red[j + s]; __syncthreads(); }
  float mean = red[0] * (1.0f / 128.0f); __syncthreads();
  float dc = acc - mean;
  red[j] = dc * dc; __syncthreads();
  for (int s = 64; s > 0; s >>= 1) { if (j < s) red[j] += red[j + s]; __syncthreads(); }
  float var = red[0] * (1.0f / 128.0f);
  float y = dc * rsqrtf(var + LN_EPS) * g[j] + b[j];
  num[gb * 128 + j] = y > 0.0f ? y : 0.0f;
}

// h2 = relu(ln(concat(num,xg) @ cls_W1)); logits = h2 @ cls_W2 + b2
__global__ void k_cls(const float* __restrict__ num, const float* __restrict__ xg,
                      const float* __restrict__ W1, const float* __restrict__ g,
                      const float* __restrict__ b, const float* __restrict__ W2,
                      const float* __restrict__ b2, float* __restrict__ logits) {
  __shared__ float feat[384];
  __shared__ float red[256];
  int gb = blockIdx.x, j = threadIdx.x;
  for (int k = j; k < 384; k += 256)
    feat[k] = (k < 128) ? num[gb * 128 + k] : xg[(size_t)gb * Dd + (k - 128)];
  __syncthreads();
  float acc = 0.0f;
  for (int k = 0; k < 384; k++) acc += feat[k] * W1[k * 256 + j];
  red[j] = acc; __syncthreads();
  for (int s = 128; s > 0; s >>= 1) { if (j < s) red[j] += red[j + s]; __syncthreads(); }
  float mean = red[0] * (1.0f / 256.0f); __syncthreads();
  float dc = acc - mean;
  red[j] = dc * dc; __syncthreads();
  for (int s = 128; s > 0; s >>= 1) { if (j < s) red[j] += red[j + s]; __syncthreads(); }
  float var = red[0] * (1.0f / 256.0f); __syncthreads();
  float y = dc * rsqrtf(var + LN_EPS) * g[j] + b[j];
  y = y > 0.0f ? y : 0.0f;
  red[j] = y * W2[j]; __syncthreads();
  for (int s = 128; s > 0; s >>= 1) { if (j < s) red[j] += red[j + s]; __syncthreads(); }
  if (j == 0) logits[gb] = red[0] + b2[0];
}

// ---------------------------------------------------------------------------
// Host launch
// ---------------------------------------------------------------------------
extern "C" void kernel_launch(void* const* d_in, const int* in_sizes, int n_in,
                              void* d_out, int out_size, void* d_ws, size_t ws_size,
                              hipStream_t stream) {
  (void)in_sizes; (void)n_in; (void)out_size; (void)ws_size;
  // ---- inputs (setup_inputs dict order) ----
  const int*   x_idx      = (const int*)  d_in[0];
  const int*   edge_index = (const int*)  d_in[1];
  const int*   oei        = (const int*)  d_in[2];
  const float* demo       = (const float*)d_in[5];
  const float* emb        = (const float*)d_in[6];
  const float* g_Wl  = (const float*)d_in[7];   const float* g_bl  = (const float*)d_in[8];
  const float* g_Wr  = (const float*)d_in[9];   const float* g_br  = (const float*)d_in[10];
  const float* g_att = (const float*)d_in[11];  const float* g_bias= (const float*)d_in[12];
  const float* ln1_g = (const float*)d_in[13];  const float* ln1_b = (const float*)d_in[14];
  const float* lin1_W= (const float*)d_in[15];  const float* lin1_b= (const float*)d_in[16];
  const float* param_W=(const float*)d_in[17];  const float* param_b=(const float*)d_in[18];
  const float* dec_Wl= (const float*)d_in[19];  const float* dec_bl= (const float*)d_in[20];
  const float* dec_Wr= (const float*)d_in[21];  const float* dec_br= (const float*)d_in[22];
  const float* dec_att=(const float*)d_in[23];  const float* dec_bias=(const float*)d_in[24];
  const float* dln_g = (const float*)d_in[25];  const float* dln_b = (const float*)d_in[26];
  const float* num_W = (const float*)d_in[27];
  const float* numln_g=(const float*)d_in[28];  const float* numln_b=(const float*)d_in[29];
  const float* cls_W1= (const float*)d_in[30];
  const float* clsln_g=(const float*)d_in[31];  const float* clsln_b=(const float*)d_in[32];
  const float* cls_W2= (const float*)d_in[33];  const float* cls_b2= (const float*)d_in[34];

  float* out_f = (float*)d_out;  // [0..127] logits, [128] kld

  // ---- workspace carve-up ----
  char* base = (char*)d_ws; size_t off = 0;
  auto carve = [&](size_t bytes) -> void* {
    void* p = base + off;
    off = (off + bytes + 255) & ~(size_t)255;
    return p;
  };
  __bf16* gWl_t  = (__bf16*)carve((size_t)Dd * Dd * 2);
  __bf16* gWr_t  = (__bf16*)carve((size_t)Dd * Dd * 2);
  __bf16* lin1_t = (__bf16*)carve((size_t)Dd * Dd * 2);
  __bf16* param_t= (__bf16*)carve((size_t)Dd * 2 * Dd * 2);
  __bf16* dWl_t  = (__bf16*)carve((size_t)Dd * Dd * 2);
  __bf16* dWr_t  = (__bf16*)carve((size_t)Dd * Dd * 2);
  __bf16* xb     = (__bf16*)carve((size_t)Nn * Dd * 2);   // bf16 activations (reused)
  __bf16* hb     = (__bf16*)carve((size_t)Nn * Dd * 2);   // bf16 lin1 output
  float*  xl     = (float*) carve((size_t)Nn * Dd * 4);
  float*  xr     = (float*) carve((size_t)Nn * Dd * 4);
  float*  outb   = (float*) carve((size_t)Nn * Dd * 4);   // GAT aggregation
  float*  hbuf   = (float*) carve((size_t)Nn * 2 * Dd * 4);
  int nE1 = Ee + Nn, nE2 = Nn + Nn;
  int*    srcb   = (int*)   carve((size_t)nE1 * 4);
  int*    dstb   = (int*)   carve((size_t)nE1 * 4);
  float*  ebuf   = (float*) carve((size_t)nE1 * 4);
  float*  exbuf  = (float*) carve((size_t)nE1 * 4);
  float*  emax   = (float*) carve((size_t)Nn * 4);
  float*  den    = (float*) carve((size_t)Nn * 4);
  float*  xg     = (float*) carve((size_t)Bb * Dd * 4);
  float*  numb   = (float*) carve((size_t)Bb * 128 * 4);

  auto cdiv = [](int a, int b) { return (a + b - 1) / b; };
  const int T = 256;

  // ---- weight conversion (fp32 -> bf16, transposed for WMMA B loads) ----
  k_w_to_bf16_t<<<cdiv(Dd * Dd, T), T, 0, stream>>>(g_Wl,   gWl_t,  Dd, Dd);
  k_w_to_bf16_t<<<cdiv(Dd * Dd, T), T, 0, stream>>>(g_Wr,   gWr_t,  Dd, Dd);
  k_w_to_bf16_t<<<cdiv(Dd * Dd, T), T, 0, stream>>>(lin1_W, lin1_t, Dd, Dd);
  k_w_to_bf16_t<<<cdiv(Dd * 2 * Dd, T), T, 0, stream>>>(param_W, param_t, Dd, 2 * Dd);
  k_w_to_bf16_t<<<cdiv(Dd * Dd, T), T, 0, stream>>>(dec_Wl, dWl_t,  Dd, Dd);
  k_w_to_bf16_t<<<cdiv(Dd * Dd, T), T, 0, stream>>>(dec_Wr, dWr_t,  Dd, Dd);

  // ---- embedding gather ----
  k_gather<<<cdiv(Nn * Dd, T), T, 0, stream>>>(x_idx, emb, xb);

  // ---- encode GATv2 ----
  int blocks_DD  = (Nn / 64) * (Dd / 64);        // 64x64 tile per block
  int blocks_D2D = (Nn / 64) * ((2 * Dd) / 64);
  k_wmma_gemm<<<blocks_DD, 128, 0, stream>>>(xb, gWl_t, g_bl, xl, nullptr, Nn, Dd, Dd);
  k_wmma_gemm<<<blocks_DD, 128, 0, stream>>>(xb, gWr_t, g_br, xr, nullptr, Nn, Dd, Dd);
  k_build_edges<<<cdiv(nE1, T), T, 0, stream>>>(edge_index, Ee, srcb, dstb);
  k_init_softmax<<<cdiv(Nn, T), T, 0, stream>>>(emax, den);
  k_init_out_bias<<<cdiv(Nn * Dd, T), T, 0, stream>>>(outb, g_bias);
  k_edge_score<<<cdiv(nE1, 8), T, 0, stream>>>(srcb, dstb, xl, xr, g_att, ebuf, emax, nE1);
  k_edge_exp<<<cdiv(nE1, T), T, 0, stream>>>(dstb, ebuf, emax, den, exbuf, nE1);
  k_edge_aggr<<<cdiv(nE1, 8), T, 0, stream>>>(srcb, dstb, xl, exbuf, den, outb, nE1);

  // ---- LN+ReLU -> lin1 -> param head ----
  k_ln_relu<<<cdiv(Nn, 4), 128, 0, stream>>>(outb, ln1_g, ln1_b, xb, Nn);
  k_wmma_gemm<<<blocks_DD, 128, 0, stream>>>(xb, lin1_t, lin1_b, nullptr, hb, Nn, Dd, Dd);
  k_wmma_gemm<<<blocks_D2D, 128, 0, stream>>>(hb, param_t, param_b, hbuf, nullptr, Nn, 2 * Dd, Dd);

  // ---- mu / KL ----
  k_zero1<<<1, 1, 0, stream>>>(out_f + Bb);
  k_mu_kl<<<cdiv(Nn, 4), 128, 0, stream>>>(hbuf, xb /* mu bf16 */, out_f + Bb);

  // ---- decode GATv2 on mu ----
  k_wmma_gemm<<<blocks_DD, 128, 0, stream>>>(xb, dWl_t, dec_bl, xl, nullptr, Nn, Dd, Dd);
  k_wmma_gemm<<<blocks_DD, 128, 0, stream>>>(xb, dWr_t, dec_br, xr, nullptr, Nn, Dd, Dd);
  k_build_edges<<<cdiv(nE2, T), T, 0, stream>>>(oei, Nn, srcb, dstb);
  k_init_softmax<<<cdiv(Nn, T), T, 0, stream>>>(emax, den);
  k_init_out_bias<<<cdiv(Nn * Dd, T), T, 0, stream>>>(outb, dec_bias);
  k_edge_score<<<cdiv(nE2, 8), T, 0, stream>>>(srcb, dstb, xl, xr, dec_att, ebuf, emax, nE2);
  k_edge_exp<<<cdiv(nE2, T), T, 0, stream>>>(dstb, ebuf, emax, den, exbuf, nE2);
  k_edge_aggr<<<cdiv(nE2, 8), T, 0, stream>>>(srcb, dstb, xl, exbuf, den, outb, nE2);

  // ---- readout + classifier ----
  k_ln_rows<<<cdiv(Bb, 4), 128, 0, stream>>>(outb, dln_g, dln_b, xg);
  k_num<<<Bb, 128, 0, stream>>>(demo, num_W, numln_g, numln_b, numb);
  k_cls<<<Bb, 256, 0, stream>>>(numb, xg, cls_W1, clsln_g, clsln_b, cls_W2, cls_b2, out_f);
}
